// Processor_31842887532968
// MI455X (gfx1250) — matrix operand order repaired and checked by reading, not compiled, not run on hardware
//
#include <hip/hip_runtime.h>
#include <hip/hip_bf16.h>

// ---------------------------------------------------------------------------
// MI455X (gfx1250) fused GNN message-passing network.
// bf16 WMMA (v_wmma_f32_16x16x32_bf16) GEMM core, fp32 accumulate,
// LDS-staged activation tiles, shuffle-based LayerNorm, fused fp32 atomic
// scatter for segment_sum. All activations + weights are L2-resident.
// waves_per_eu(4) caps VGPRs at <=256/wave: no vgpr-msb addressing, no
// scratch spills, 4 waves/SIMD occupancy to hide L2/LDS latency.
// ---------------------------------------------------------------------------

typedef __attribute__((ext_vector_type(16))) __bf16 v16bf;
typedef __attribute__((ext_vector_type(8)))  __bf16 v8bf;
typedef __attribute__((ext_vector_type(8)))  float  v8f;

union ABReg { v16bf v; v8bf h[2]; };

#define DEV static __device__ __forceinline__

DEV v8f wmma_bf16(v16bf a, v16bf b, v8f c) {
  // 8 args: (neg_a, A, neg_b, B, c_mod, C, reuse_a, reuse_b)
  return __builtin_amdgcn_wmma_f32_16x16x32_bf16(false, a, false, b,
                                                 (short)0, c, false, false);
}

DEV void zacc(v8f acc[2][2]) {
#pragma unroll
  for (int i = 0; i < 2; ++i)
#pragma unroll
    for (int j = 0; j < 2; ++j)
#pragma unroll
      for (int k = 0; k < 8; ++k) acc[i][j][k] = 0.0f;
}

// GEMM core: A tile [32 x ksteps*32] bf16 in LDS (row-major, stride lda),
// B = transposed weight wT [256][ldw] bf16 in global (L2-hot).
// Each wave computes a 32x32 output tile: acc[mt][nt] (16x16 each).
DEV void gemm_tiles(const __bf16* __restrict__ sA, int lda,
                    const __bf16* __restrict__ wT, int ldw,
                    int kB0, int ksteps,
                    v8f acc[2][2], int lane, int wave)
{
  const int hi = lane >> 4, lo = lane & 15;
#pragma unroll 2
  for (int ks = 0; ks < ksteps; ++ks) {
    ABReg b0, b1, a0, a1;
    // B fragment: lane holds 16 consecutive K of column n (contiguous in wT)
    const int kb = kB0 + ks * 32 + hi * 16;
    const __bf16* p0 = wT + (size_t)(wave * 32 + lo) * ldw + kb;
    b0.h[0] = *(const v8bf*)(p0);
    b0.h[1] = *(const v8bf*)(p0 + 8);
    const __bf16* p1 = p0 + (size_t)16 * ldw;
    b1.h[0] = *(const v8bf*)(p1);
    b1.h[1] = *(const v8bf*)(p1 + 8);
    // A fragment: lane holds 16 consecutive K [ka..ka+16) of its row
    const int ka = ks * 32 + hi * 16;
    const __bf16* ap0 = sA + (size_t)lo * lda + ka;
    a0.h[0] = *(const v8bf*)(ap0);
    a0.h[1] = *(const v8bf*)(ap0 + 8);
    const __bf16* ap1 = ap0 + (size_t)16 * lda;
    a1.h[0] = *(const v8bf*)(ap1);
    a1.h[1] = *(const v8bf*)(ap1 + 8);
    acc[0][0] = wmma_bf16(a0.v, b0.v, acc[0][0]);
    acc[0][1] = wmma_bf16(a0.v, b1.v, acc[0][1]);
    acc[1][0] = wmma_bf16(a1.v, b0.v, acc[1][0]);
    acc[1][1] = wmma_bf16(a1.v, b1.v, acc[1][1]);
  }
}

// C/D layout: VGPR g -> (m = mt*16 + 8*(lane>>4) + g, n = base + (lane&15))
DEV void store_hidden(v8f acc[2][2], const float* __restrict__ bias,
                      __bf16* __restrict__ sH, int lane, int wave, bool relu)
{
  const int hi = lane >> 4, lo = lane & 15;
#pragma unroll
  for (int mt = 0; mt < 2; ++mt)
#pragma unroll
    for (int nt = 0; nt < 2; ++nt) {
      const int n = wave * 32 + nt * 16 + lo;
      const float bv = bias[n];
      const int mbase = mt * 16 + hi * 8;
#pragma unroll
      for (int g = 0; g < 8; ++g) {
        float v = acc[mt][nt][g] + bv;
        if (relu) v = fmaxf(v, 0.0f);
        sH[(size_t)(mbase + g) * 264 + n] = (__bf16)v;
      }
    }
}

DEV void store_f32(v8f acc[2][2], const float* __restrict__ bias,
                   float* __restrict__ sF, int lane, int wave)
{
  const int hi = lane >> 4, lo = lane & 15;
#pragma unroll
  for (int mt = 0; mt < 2; ++mt)
#pragma unroll
    for (int nt = 0; nt < 2; ++nt) {
      const int n = wave * 32 + nt * 16 + lo;
      const float bv = bias[n];
      const int mbase = mt * 16 + hi * 8;
#pragma unroll
      for (int g = 0; g < 8; ++g)
        sF[(size_t)(mbase + g) * 260 + n] = acc[mt][nt][g] + bv;
    }
}

// ---------------------------------------------------------------------------
// Edge kernel: ea += MLP([x[src], x[dst], ea]); scatter-add into agg[dst].
// LDS plan (50176 B): A-half tile [32][392]bf16 @0 (25088, reused as H2),
// H1 [32][264]bf16 @25088 (16896), F32 out [32][260]f32 @16896 (33280).
// ---------------------------------------------------------------------------
__global__ __launch_bounds__(256)
__attribute__((amdgpu_waves_per_eu(4)))
void edge_mlp_kernel(
    const __bf16* __restrict__ xbf, __bf16* __restrict__ eabf,
    float* __restrict__ eacur, const int* __restrict__ ei,
    const __bf16* __restrict__ w1, const __bf16* __restrict__ w2,
    const __bf16* __restrict__ w3,
    const float* __restrict__ b1, const float* __restrict__ b2,
    const float* __restrict__ b3,
    const float* __restrict__ lng, const float* __restrict__ lnb,
    float* __restrict__ agg, int E)
{
  extern __shared__ char smem[];
  __bf16* sA  = (__bf16*)(smem);           // also reused as hidden2
  __bf16* sH1 = (__bf16*)(smem + 25088);
  float*  sF  = (float*)(smem + 16896);
  const int tid = threadIdx.x, lane = tid & 31, wave = tid >> 5;
  const int e0 = blockIdx.x * 32;

  v8f acc[2][2];
  zacc(acc);

  // ---- layer 1 (K=768) in two K-halves of 384, staged via LDS ----
#pragma unroll 1
  for (int hh = 0; hh < 2; ++hh) {
    for (int i = tid; i < 32 * 48; i += 256) {
      const int r = i / 48, c = i % 48;
      const int kg = hh * 384 + c * 8;
      const int e = e0 + r;
      const __bf16* sp;
      if (kg < 256)       sp = xbf + (size_t)ei[e] * 256 + kg;
      else if (kg < 512)  sp = xbf + (size_t)ei[E + e] * 256 + (kg - 256);
      else                sp = eabf + (size_t)e * 256 + (kg - 512);
      *(uint4*)(sA + (size_t)r * 392 + c * 8) = *(const uint4*)sp;
    }
    __syncthreads();
    gemm_tiles(sA, 392, w1, 768, hh * 384, 12, acc, lane, wave);
    __syncthreads();
  }
  store_hidden(acc, b1, sH1, lane, wave, true);
  __syncthreads();

  // ---- layer 2 (K=256): H1 -> H2 (in sA region) ----
  zacc(acc);
  gemm_tiles(sH1, 264, w2, 256, 0, 8, acc, lane, wave);
  store_hidden(acc, b2, sA, lane, wave, true);
  __syncthreads();

  // ---- layer 3 (K=256): H2 -> fp32 out ----
  zacc(acc);
  gemm_tiles(sA, 264, w3, 256, 0, 8, acc, lane, wave);
  store_f32(acc, b3, sF, lane, wave);
  __syncthreads();

  // ---- LayerNorm + residual + fused scatter-add (4 rows per wave) ----
#pragma unroll 1
  for (int rr = 0; rr < 4; ++rr) {
    const int r = wave * 4 + rr;
    float hv[8], s = 0.0f, s2 = 0.0f;
#pragma unroll
    for (int j = 0; j < 8; ++j) {
      hv[j] = sF[(size_t)r * 260 + lane + j * 32];
      s += hv[j]; s2 += hv[j] * hv[j];
    }
#pragma unroll
    for (int o = 16; o > 0; o >>= 1) {
      s  += __shfl_xor(s,  o, 32);
      s2 += __shfl_xor(s2, o, 32);
    }
    const float mean = s * (1.0f / 256.0f);
    const float rs = rsqrtf(fmaxf(s2 * (1.0f / 256.0f) - mean * mean, 0.0f) + 1e-5f);
    const int eg = e0 + r;
    const int dst = ei[E + eg];
#pragma unroll
    for (int j = 0; j < 8; ++j) {
      const int n = lane + j * 32;
      const size_t idx = (size_t)eg * 256 + n;
      const float val = (hv[j] - mean) * rs * lng[n] + lnb[n] + eacur[idx];
      eacur[idx] = val;
      eabf[idx]  = (__bf16)val;
      atomicAdd(agg + (size_t)dst * 256 + n, val);
    }
  }
}

// ---------------------------------------------------------------------------
// Node kernel: x += MLP([x, agg]).
// LDS plan (50176 B): A-half [32][264]bf16 @0 (16896, reused as H2),
// H1 @16896 (16896), F32 out @16896 (33280, after H1 is dead).
// ---------------------------------------------------------------------------
__global__ __launch_bounds__(256)
__attribute__((amdgpu_waves_per_eu(4)))
void node_mlp_kernel(
    __bf16* __restrict__ xbf, const float* __restrict__ agg,
    float* __restrict__ xcur,
    const __bf16* __restrict__ w1, const __bf16* __restrict__ w2,
    const __bf16* __restrict__ w3,
    const float* __restrict__ b1, const float* __restrict__ b2,
    const float* __restrict__ b3,
    const float* __restrict__ lng, const float* __restrict__ lnb, int N)
{
  extern __shared__ char smem[];
  __bf16* sA  = (__bf16*)(smem);
  __bf16* sH1 = (__bf16*)(smem + 16896);
  float*  sF  = (float*)(smem + 16896);
  const int tid = threadIdx.x, lane = tid & 31, wave = tid >> 5;
  const int n0 = blockIdx.x * 32;

  v8f acc[2][2];
  zacc(acc);

  // ---- layer 1 half 0: x (bf16 copy) ----
  for (int i = tid; i < 32 * 32; i += 256) {
    const int r = i >> 5, c = i & 31;
    int node = n0 + r; if (node >= N) node = N - 1;
    *(uint4*)(sA + (size_t)r * 264 + c * 8) =
        *(const uint4*)(xbf + (size_t)node * 256 + c * 8);
  }
  __syncthreads();
  gemm_tiles(sA, 264, w1, 512, 0, 8, acc, lane, wave);
  __syncthreads();

  // ---- layer 1 half 1: agg (fp32 -> bf16 on stage) ----
  for (int i = tid; i < 32 * 32; i += 256) {
    const int r = i >> 5, c = i & 31;
    int node = n0 + r; if (node >= N) node = N - 1;
    const float* fp = agg + (size_t)node * 256 + c * 8;
    v8bf t;
#pragma unroll
    for (int j = 0; j < 8; ++j) t[j] = (__bf16)fp[j];
    *(v8bf*)(sA + (size_t)r * 264 + c * 8) = t;
  }
  __syncthreads();
  gemm_tiles(sA, 264, w1, 512, 256, 8, acc, lane, wave);
  store_hidden(acc, b1, sH1, lane, wave, true);
  __syncthreads();

  zacc(acc);
  gemm_tiles(sH1, 264, w2, 256, 0, 8, acc, lane, wave);
  store_hidden(acc, b2, sA, lane, wave, true);
  __syncthreads();

  zacc(acc);
  gemm_tiles(sA, 264, w3, 256, 0, 8, acc, lane, wave);
  store_f32(acc, b3, sF, lane, wave);
  __syncthreads();

#pragma unroll 1
  for (int rr = 0; rr < 4; ++rr) {
    const int r = wave * 4 + rr;
    float hv[8], s = 0.0f, s2 = 0.0f;
#pragma unroll
    for (int j = 0; j < 8; ++j) {
      hv[j] = sF[(size_t)r * 260 + lane + j * 32];
      s += hv[j]; s2 += hv[j] * hv[j];
    }
#pragma unroll
    for (int o = 16; o > 0; o >>= 1) {
      s  += __shfl_xor(s,  o, 32);
      s2 += __shfl_xor(s2, o, 32);
    }
    const float mean = s * (1.0f / 256.0f);
    const float rs = rsqrtf(fmaxf(s2 * (1.0f / 256.0f) - mean * mean, 0.0f) + 1e-5f);
    const int node = n0 + r;
    if (node < N) {
#pragma unroll
      for (int j = 0; j < 8; ++j) {
        const int n = lane + j * 32;
        const size_t idx = (size_t)node * 256 + n;
        const float val = (hv[j] - mean) * rs * lng[n] + lnb[n] + xcur[idx];
        xcur[idx] = val;
        xbf[idx]  = (__bf16)val;
      }
    }
  }
}

// ---------------------------------------------------------------------------
// Prep kernels
// ---------------------------------------------------------------------------
// W: [9][K][256] fp32 -> out: [9][256][K] bf16 (transposed per block)
__global__ void wconv_kernel(const float* __restrict__ W,
                             __bf16* __restrict__ out, int K, size_t total)
{
  const size_t i = (size_t)blockIdx.x * 256 + threadIdx.x;
  if (i >= total) return;
  const size_t per = (size_t)K * 256;
  const int b = (int)(i / per);
  const size_t rem = i - (size_t)b * per;
  const int n = (int)(rem / K);
  const int k = (int)(rem - (size_t)n * K);
  out[i] = (__bf16)W[((size_t)b * K + k) * 256 + n];
}

__global__ void cvt_copy_kernel(const float* __restrict__ in,
                                float* __restrict__ fo,
                                __bf16* __restrict__ bo, size_t n)
{
  const size_t i = (size_t)blockIdx.x * 256 + threadIdx.x;
  if (i < n) { const float v = in[i]; fo[i] = v; bo[i] = (__bf16)v; }
}

__global__ void zero_kernel(float* __restrict__ p, size_t n)
{
  const size_t i = (size_t)blockIdx.x * 256 + threadIdx.x;
  if (i < n) p[i] = 0.0f;
}

// ---------------------------------------------------------------------------
extern "C" void kernel_launch(void* const* d_in, const int* in_sizes, int n_in,
                              void* d_out, int out_size, void* d_ws, size_t ws_size,
                              hipStream_t stream)
{
  (void)in_sizes; (void)n_in; (void)out_size; (void)ws_size;
  const int N = 10000, E = 60000, D = 256, B = 9;

  const float* x    = (const float*)d_in[0];
  const int*   ei   = (const int*)  d_in[1];
  const float* ea   = (const float*)d_in[2];
  const float* eW1  = (const float*)d_in[3];
  const float* eB1  = (const float*)d_in[4];
  const float* eW2  = (const float*)d_in[5];
  const float* eB2  = (const float*)d_in[6];
  const float* eW3  = (const float*)d_in[7];
  const float* eB3  = (const float*)d_in[8];
  const float* eLNg = (const float*)d_in[9];
  const float* eLNb = (const float*)d_in[10];
  const float* nW1  = (const float*)d_in[11];
  const float* nB1  = (const float*)d_in[12];
  const float* nW2  = (const float*)d_in[13];
  const float* nB2  = (const float*)d_in[14];
  const float* nW3  = (const float*)d_in[15];
  const float* nB3  = (const float*)d_in[16];
  const float* nLNg = (const float*)d_in[17];
  const float* nLNb = (const float*)d_in[18];

  char* ws = (char*)d_ws;
  size_t off = 0;
  auto take = [&](size_t bytes) -> char* {
    char* p = ws + off;
    off += (bytes + 255) & ~(size_t)255;
    return p;
  };
  float*  xcur  = (float*) take((size_t)N * D * 4);
  __bf16* xbf   = (__bf16*)take((size_t)N * D * 2);
  float*  eacur = (float*) take((size_t)E * D * 4);
  __bf16* eabf  = (__bf16*)take((size_t)E * D * 2);
  float*  agg   = (float*) take((size_t)N * D * 4);
  __bf16* wTe1  = (__bf16*)take((size_t)B * 768 * 256 * 2);
  __bf16* wTe2  = (__bf16*)take((size_t)B * 256 * 256 * 2);
  __bf16* wTe3  = (__bf16*)take((size_t)B * 256 * 256 * 2);
  __bf16* wTn1  = (__bf16*)take((size_t)B * 512 * 256 * 2);
  __bf16* wTn2  = (__bf16*)take((size_t)B * 256 * 256 * 2);
  __bf16* wTn3  = (__bf16*)take((size_t)B * 256 * 256 * 2);

  // ---- one-time: transpose+cast all weights to bf16 [B][256][K] ----
  auto wconv = [&](const float* W, __bf16* o, int K) {
    const size_t total = (size_t)B * 256 * K;
    wconv_kernel<<<dim3((unsigned)((total + 255) / 256)), dim3(256), 0, stream>>>(
        W, o, K, total);
  };
  wconv(eW1, wTe1, 768);
  wconv(eW2, wTe2, 256);
  wconv(eW3, wTe3, 256);
  wconv(nW1, wTn1, 512);
  wconv(nW2, wTn2, 256);
  wconv(nW3, wTn3, 256);

  // ---- init: fp32 working copies + bf16 shadows of x and edge_attr ----
  cvt_copy_kernel<<<dim3((N * D + 255) / 256), dim3(256), 0, stream>>>(
      x, xcur, xbf, (size_t)N * D);
  cvt_copy_kernel<<<dim3((E * D + 255) / 256), dim3(256), 0, stream>>>(
      ea, eacur, eabf, (size_t)E * D);

  const size_t SMEM = 50176;
  for (int b = 0; b < B; ++b) {
    zero_kernel<<<dim3((N * D + 255) / 256), dim3(256), 0, stream>>>(
        agg, (size_t)N * D);
    edge_mlp_kernel<<<dim3(E / 32), dim3(256), SMEM, stream>>>(
        xbf, eabf, eacur, ei,
        wTe1 + (size_t)b * 768 * 256,
        wTe2 + (size_t)b * 256 * 256,
        wTe3 + (size_t)b * 256 * 256,
        eB1 + b * 256, eB2 + b * 256, eB3 + b * 256,
        eLNg + b * 256, eLNb + b * 256, agg, E);
    node_mlp_kernel<<<dim3((N + 31) / 32), dim3(256), SMEM, stream>>>(
        xbf, agg, xcur,
        wTn1 + (size_t)b * 512 * 256,
        wTn2 + (size_t)b * 256 * 256,
        wTn3 + (size_t)b * 256 * 256,
        nB1 + b * 256, nB2 + b * 256, nB3 + b * 256,
        nLNg + b * 256, nLNb + b * 256, N);
  }

  hipMemcpyAsync(d_out, xcur, (size_t)N * D * 4, hipMemcpyDeviceToDevice, stream);
}
